// Frame_Layer_31164282700653
// MI455X (gfx1250) — matrix operand order repaired and checked by reading, not compiled, not run on hardware
//
#include <hip/hip_runtime.h>

typedef _Float16 half8  __attribute__((ext_vector_type(8)));
typedef _Float16 half16 __attribute__((ext_vector_type(16)));
typedef float    float8 __attribute__((ext_vector_type(8)));

union H16 { half16 v; half8 h[2]; };

__device__ __forceinline__ float8 wmma_f16(half16 a, half16 b, float8 c) {
  // D = A(16x32 f16) * B(32x16 f16) + C(16x16 f32)
  return __builtin_amdgcn_wmma_f32_16x16x32_f16(false, a, false, b, (short)0, c,
                                                false, false);
}

// ---------------- elementwise helpers ----------------
__global__ __launch_bounds__(256) void cvt_kernel(const float* __restrict__ in,
                                                  _Float16* __restrict__ out, int n) {
  int i = blockIdx.x * 256 + threadIdx.x;
  if (i < n) out[i] = (_Float16)in[i];
}

__global__ __launch_bounds__(256) void zero_kernel(float* __restrict__ p, int n) {
  int i = blockIdx.x * 256 + threadIdx.x;
  if (i < n) p[i] = 0.0f;
}

// ---------------- LayerNorm over D=512, one wave per row ----------------
__global__ __launch_bounds__(256) void ln_kernel(const float* __restrict__ X,
                                                 const float* __restrict__ w,
                                                 const float* __restrict__ bvec,
                                                 float* __restrict__ O32,
                                                 _Float16* __restrict__ O16, int N) {
  int row = blockIdx.x * 8 + (threadIdx.x >> 5);
  if (row >= N) return;
  int lane = threadIdx.x & 31;
  const float* x = X + (size_t)row * 512;
  float vals[16];
  float s = 0.0f;
#pragma unroll
  for (int i = 0; i < 16; ++i) { vals[i] = x[lane + i * 32]; s += vals[i]; }
#pragma unroll
  for (int off = 16; off > 0; off >>= 1) s += __shfl_xor(s, off, 32);
  float mu = s * (1.0f / 512.0f);
  float vs = 0.0f;
#pragma unroll
  for (int i = 0; i < 16; ++i) { float d = vals[i] - mu; vs += d * d; }
#pragma unroll
  for (int off = 16; off > 0; off >>= 1) vs += __shfl_xor(vs, off, 32);
  float rstd = rsqrtf(vs * (1.0f / 512.0f) + 1e-5f);
#pragma unroll
  for (int i = 0; i < 16; ++i) {
    int c = lane + i * 32;
    float y = (vals[i] - mu) * rstd * w[c] + bvec[c];
    if (O32) O32[(size_t)row * 512 + c] = y;
    if (O16) O16[(size_t)row * 512 + c] = (_Float16)y;
  }
}

// ---------------- NT GEMM: C[N,DOUT] = A[N,K] * W[DOUT,K]^T (+bias,+resid,relu) ----
// Wave computes a 16x64 C block (A tile reused over 4 WMMAs per K step).
__global__ __launch_bounds__(256) void gemm_nt(const _Float16* __restrict__ A,
                                               const _Float16* __restrict__ W,
                                               const float* __restrict__ bias,
                                               float* __restrict__ C32,
                                               _Float16* __restrict__ C16,
                                               const float* __restrict__ resid,
                                               int N, int K, int DOUT, int relu,
                                               float oscale) {
  int wid = blockIdx.x * 8 + (threadIdx.x >> 5);
  int nt4cnt = DOUT >> 6;
  int mt = wid / nt4cnt;
  int nt4 = wid - mt * nt4cnt;
  if (mt >= (N >> 4)) return;
  int lane = threadIdx.x & 31;
  int r = lane & 15, hi = lane >> 4;
  const _Float16* arow = A + (size_t)(mt * 16 + r) * K;
  const _Float16* w0 = W + (size_t)(nt4 * 64 + r) * K;
  const _Float16* w1 = w0 + (size_t)16 * K;
  const _Float16* w2 = w0 + (size_t)32 * K;
  const _Float16* w3 = w0 + (size_t)48 * K;
  float8 acc0 = {}, acc1 = {}, acc2 = {}, acc3 = {};
  for (int k0 = 0; k0 < K; k0 += 32) {
    H16 a, b0, b1, b2, b3;
    a.h[0] = *(const half8*)(arow + k0 + hi * 16);
    a.h[1] = *(const half8*)(arow + k0 + hi * 16 + 8);
    int kb = k0 + hi * 16;
    b0.h[0] = *(const half8*)(w0 + kb); b0.h[1] = *(const half8*)(w0 + kb + 8);
    b1.h[0] = *(const half8*)(w1 + kb); b1.h[1] = *(const half8*)(w1 + kb + 8);
    b2.h[0] = *(const half8*)(w2 + kb); b2.h[1] = *(const half8*)(w2 + kb + 8);
    b3.h[0] = *(const half8*)(w3 + kb); b3.h[1] = *(const half8*)(w3 + kb + 8);
    acc0 = wmma_f16(a.v, b0.v, acc0);
    acc1 = wmma_f16(a.v, b1.v, acc1);
    acc2 = wmma_f16(a.v, b2.v, acc2);
    acc3 = wmma_f16(a.v, b3.v, acc3);
  }
  float8 accs[4] = {acc0, acc1, acc2, acc3};
#pragma unroll
  for (int j = 0; j < 4; ++j) {
    int col = nt4 * 64 + j * 16 + r;
    float bv = bias[col];
#pragma unroll
    for (int v = 0; v < 8; ++v) {
      int m = mt * 16 + v + 8 * hi;
      float val = accs[j][v] + bv;
      if (resid) val += resid[(size_t)m * DOUT + col];
      if (relu) val = fmaxf(val, 0.0f);
      if (C32) C32[(size_t)m * DOUT + col] = val;
      if (C16) C16[(size_t)m * DOUT + col] = (_Float16)(val * oscale);
    }
  }
}

// ---------------- V transpose: [Lk,B,H*64] -> [B,H,64,Lk] ----------------
__global__ __launch_bounds__(256) void vtrans_kernel(const _Float16* __restrict__ vsrc,
                                                     _Float16* __restrict__ vt) {
  int idx = blockIdx.x * 256 + threadIdx.x;  // flat over (b,h,dh,k), k fastest
  int k = idx & 4095;
  int t = idx >> 12;
  int dh = t & 63;
  int t2 = t >> 6;
  int h = t2 & 7;
  int b = t2 >> 3;
  vt[idx] = vsrc[((size_t)(k * 4 + b)) * 512 + h * 64 + dh];
}

// ---------------- Fused attention: scores + mask + softmax + mean + P*V --------
// One block (8 waves) per (b, h, 16-query tile). 16x4096 f32 score block in LDS.
__global__ __launch_bounds__(256) void attn_kernel(const _Float16* __restrict__ q16,
                                                   const _Float16* __restrict__ k16,
                                                   const _Float16* __restrict__ vt16,
                                                   _Float16* __restrict__ ctx16,
                                                   float* __restrict__ attn_out) {
  extern __shared__ float S[];  // [16][4096]
  __shared__ float red[16][16];
  __shared__ float rowstat[16];
  __shared__ float cp[8][256];

  const int bid = blockIdx.x;
  const int qt = bid & 31;
  const int h = (bid >> 5) & 7;
  const int b = bid >> 8;
  const int tid = threadIdx.x;
  const int wave = tid >> 5;
  const int lane = tid & 31;
  const int r = lane & 15;
  const int hi = lane >> 4;

  // ---- Phase A: S = (q*scale) @ k^T with adaptive mask (scale folded into q16) --
  const _Float16* qrow = q16 + ((size_t)((qt * 16 + r) * 4 + b)) * 512 + h * 64;
  H16 a0, a1;
  a0.h[0] = *(const half8*)(qrow + hi * 16);
  a0.h[1] = *(const half8*)(qrow + hi * 16 + 8);
  a1.h[0] = *(const half8*)(qrow + 32 + hi * 16);
  a1.h[1] = *(const half8*)(qrow + 32 + hi * 16 + 8);

  for (int i = 0; i < 32; ++i) {
    const int kt = wave + (i << 3);  // 256 k-tiles interleaved across 8 waves
    const _Float16* krow = k16 + ((size_t)((kt * 16 + r) * 4 + b)) * 512 + h * 64;
    H16 b0, b1;
    b0.h[0] = *(const half8*)(krow + hi * 16);
    b0.h[1] = *(const half8*)(krow + hi * 16 + 8);
    b1.h[0] = *(const half8*)(krow + 32 + hi * 16);
    b1.h[1] = *(const half8*)(krow + 32 + hi * 16 + 8);
    float8 acc = {};
    acc = wmma_f16(a0.v, b0.v, acc);
    acc = wmma_f16(a1.v, b1.v, acc);
    const int kcol = kt * 16 + r;
#pragma unroll
    for (int v = 0; v < 8; ++v) {
      const int ql = v + 8 * hi;
      const int qg = qt * 16 + ql;
      // adaptive mask: allowed cols [s, e); aa=512, bb=4096, ada=0.2
      int s = qg * 8, e = s + 827;
      if (e > 4095) { s = 3268; e = 4096; }
      S[ql * 4096 + kcol] = (kcol >= s && kcol < e) ? acc[v] : -1.0e30f;
    }
  }
  __syncthreads();

  // ---- softmax per row + mean-over-heads accumulation into d_out ----
  {
    const int row = tid >> 4;   // 16 rows x 16 threads
    const int sub = tid & 15;
    float* Srow = S + row * 4096;
    float mx = -1.0e30f;
    for (int c = sub; c < 4096; c += 16) mx = fmaxf(mx, Srow[c]);
    red[row][sub] = mx;
    __syncthreads();
    if (sub == 0) {
      float m = red[row][0];
      for (int j = 1; j < 16; ++j) m = fmaxf(m, red[row][j]);
      rowstat[row] = m;
    }
    __syncthreads();
    const float rowmax = rowstat[row];
    float sum = 0.0f;
    for (int c = sub; c < 4096; c += 16) {
      float ev = __expf(Srow[c] - rowmax);
      Srow[c] = ev;
      sum += ev;
    }
    __syncthreads();
    red[row][sub] = sum;
    __syncthreads();
    if (sub == 0) {
      float t = 0.0f;
      for (int j = 0; j < 16; ++j) t += red[row][j];
      rowstat[row] = 1.0f / t;
    }
    __syncthreads();
    const float inv = rowstat[row];
    const int qg2 = qt * 16 + row;
    float* arow = attn_out + ((size_t)(b * 512 + qg2)) * 4096;
    for (int c = sub; c < 4096; c += 16) {
      float p = Srow[c] * inv;
      Srow[c] = p;
      atomicAdd(&arow[c], 0.125f * p);  // mean over H=8 heads
    }
  }
  __syncthreads();

  // ---- Phase B: ctx = P @ V (Vt is [B,H,64,Lk], so NT form) ----
  {
    const int nt = wave & 3;   // which 16-wide hd tile of 64
    const int kh = wave >> 2;  // K-dimension halved across wave pairs
    const _Float16* vrow =
        vt16 + ((size_t)((b * 8 + h) * 64 + nt * 16 + r)) * 4096;
    float8 acc = {};
    for (int i = 0; i < 64; ++i) {
      const int k0 = kh * 2048 + i * 32;
      const float4* s0 = (const float4*)(S + r * 4096 + k0 + hi * 8);
      const float4* s1 = (const float4*)(S + r * 4096 + k0 + 16 + hi * 8);
      float4 x0 = s0[0], x1 = s0[1], y0 = s1[0], y1 = s1[1];
      H16 a, bb;
      a.h[0][0] = (_Float16)x0.x; a.h[0][1] = (_Float16)x0.y;
      a.h[0][2] = (_Float16)x0.z; a.h[0][3] = (_Float16)x0.w;
      a.h[0][4] = (_Float16)x1.x; a.h[0][5] = (_Float16)x1.y;
      a.h[0][6] = (_Float16)x1.z; a.h[0][7] = (_Float16)x1.w;
      a.h[1][0] = (_Float16)y0.x; a.h[1][1] = (_Float16)y0.y;
      a.h[1][2] = (_Float16)y0.z; a.h[1][3] = (_Float16)y0.w;
      a.h[1][4] = (_Float16)y1.x; a.h[1][5] = (_Float16)y1.y;
      a.h[1][6] = (_Float16)y1.z; a.h[1][7] = (_Float16)y1.w;
      bb.h[0] = *(const half8*)(vrow + k0 + hi * 8);
      bb.h[1] = *(const half8*)(vrow + k0 + 16 + hi * 8);
      acc = wmma_f16(a.v, bb.v, acc);
    }
#pragma unroll
    for (int v = 0; v < 8; ++v) cp[wave][v * 32 + lane] = acc[v];
  }
  __syncthreads();
  if (wave < 4) {
#pragma unroll
    for (int v = 0; v < 8; ++v) {
      float sv = cp[wave][v * 32 + lane] + cp[wave + 4][v * 32 + lane];
      const int qg = qt * 16 + v + 8 * hi;
      const int col = h * 64 + wave * 16 + r;
      ctx16[((size_t)(qg * 4 + b)) * 512 + col] = (_Float16)sv;
    }
  }
}

// =======================================================================
extern "C" void kernel_launch(void* const* d_in, const int* in_sizes, int n_in,
                              void* d_out, int out_size, void* d_ws, size_t ws_size,
                              hipStream_t stream) {
  (void)in_sizes; (void)n_in; (void)out_size; (void)ws_size;
  const float* tgt = (const float*)d_in[0];
  const float* mem = (const float*)d_in[1];
  const float* Wq = (const float*)d_in[2];  const float* bq = (const float*)d_in[3];
  const float* Wk = (const float*)d_in[4];  const float* bk = (const float*)d_in[5];
  const float* Wv = (const float*)d_in[6];  const float* bv = (const float*)d_in[7];
  const float* Wo = (const float*)d_in[8];  const float* bo = (const float*)d_in[9];
  const float* W1 = (const float*)d_in[10]; const float* b1 = (const float*)d_in[11];
  const float* W2 = (const float*)d_in[12]; const float* b2 = (const float*)d_in[13];
  const float* ln1w = (const float*)d_in[14]; const float* ln1b = (const float*)d_in[15];
  const float* ln2w = (const float*)d_in[16]; const float* ln2b = (const float*)d_in[17];
  const float* ln3w = (const float*)d_in[18]; const float* ln3b = (const float*)d_in[19];
  const float* ln4w = (const float*)d_in[20]; const float* ln4b = (const float*)d_in[21];

  constexpr int Lq = 512, Lk = 4096, B = 4, D = 512, DFF = 2048, H = 8;
  const int Nq = Lq * B;   // 2048
  const int Nk = Lk * B;   // 16384

  char* ws = (char*)d_ws;
  size_t off = 0;
  auto alloc = [&](size_t bytes) -> void* {
    void* p = ws + off;
    off += (bytes + 255) & ~(size_t)255;
    return p;
  };
  _Float16* Wq16 = (_Float16*)alloc((size_t)D * D * 2);
  _Float16* Wk16 = (_Float16*)alloc((size_t)D * D * 2);
  _Float16* Wv16 = (_Float16*)alloc((size_t)D * D * 2);
  _Float16* Wo16 = (_Float16*)alloc((size_t)D * D * 2);
  _Float16* W116 = (_Float16*)alloc((size_t)DFF * D * 2);
  _Float16* W216 = (_Float16*)alloc((size_t)D * DFF * 2);
  float*    t32  = (float*)alloc((size_t)Nq * D * 4);
  _Float16* t16  = (_Float16*)alloc((size_t)Nq * D * 2);
  _Float16* m16  = (_Float16*)alloc((size_t)Nk * D * 2);
  _Float16* qb16 = (_Float16*)alloc((size_t)Nq * D * 2);
  _Float16* kb16 = (_Float16*)alloc((size_t)Nk * D * 2);
  _Float16* vb16 = (_Float16*)alloc((size_t)Nk * D * 2);
  _Float16* vt16 = (_Float16*)alloc((size_t)Nk * D * 2);
  _Float16* ctx16 = (_Float16*)alloc((size_t)Nq * D * 2);
  float*    x32  = (float*)alloc((size_t)Nq * D * 4);
  float*    xl32 = (float*)alloc((size_t)Nq * D * 4);
  _Float16* xl16 = (_Float16*)alloc((size_t)Nq * D * 2);
  _Float16* hb16 = (_Float16*)alloc((size_t)Nq * DFF * 2);
  float*    y32  = (float*)alloc((size_t)Nq * D * 4);

  // weights -> f16
  cvt_kernel<<<(D * D + 255) / 256, 256, 0, stream>>>(Wq, Wq16, D * D);
  cvt_kernel<<<(D * D + 255) / 256, 256, 0, stream>>>(Wk, Wk16, D * D);
  cvt_kernel<<<(D * D + 255) / 256, 256, 0, stream>>>(Wv, Wv16, D * D);
  cvt_kernel<<<(D * D + 255) / 256, 256, 0, stream>>>(Wo, Wo16, D * D);
  cvt_kernel<<<(DFF * D + 255) / 256, 256, 0, stream>>>(W1, W116, DFF * D);
  cvt_kernel<<<(D * DFF + 255) / 256, 256, 0, stream>>>(W2, W216, D * DFF);

  // t = LN1(tgt) (f32 + f16); m = LN2(memory) (f16 only)
  ln_kernel<<<Nq / 8, 256, 0, stream>>>(tgt, ln1w, ln1b, t32, t16, Nq);
  ln_kernel<<<Nk / 8, 256, 0, stream>>>(mem, ln2w, ln2b, nullptr, m16, Nk);

  auto ggrid = [](int N, int DOUT) { return ((N / 16) * (DOUT / 64) + 7) / 8; };
  // projections (q pre-scaled by 1/sqrt(hd)=0.125)
  gemm_nt<<<ggrid(Nq, D), 256, 0, stream>>>(t16, Wq16, bq, nullptr, qb16, nullptr,
                                            Nq, D, D, 0, 0.125f);
  gemm_nt<<<ggrid(Nk, D), 256, 0, stream>>>(m16, Wk16, bk, nullptr, kb16, nullptr,
                                            Nk, D, D, 0, 1.0f);
  gemm_nt<<<ggrid(Nk, D), 256, 0, stream>>>(m16, Wv16, bv, nullptr, vb16, nullptr,
                                            Nk, D, D, 0, 1.0f);
  vtrans_kernel<<<(B * H * 64 * Lk) / 256, 256, 0, stream>>>(vb16, vt16);

  // attention
  float* attn = (float*)d_out + (size_t)Nq * D;
  zero_kernel<<<(B * Lq * Lk) / 256, 256, 0, stream>>>(attn, B * Lq * Lk);
  (void)hipFuncSetAttribute((const void*)attn_kernel,
                            hipFuncAttributeMaxDynamicSharedMemorySize,
                            16 * 4096 * 4);
  attn_kernel<<<B * H * (Lq / 16), 256, 16 * 4096 * 4, stream>>>(qb16, kb16, vt16,
                                                                 ctx16, attn);

  // out proj + residual, LN3, FFN, LN4 -> d_out
  gemm_nt<<<ggrid(Nq, D), 256, 0, stream>>>(ctx16, Wo16, bo, x32, nullptr, t32,
                                            Nq, D, D, 0, 1.0f);
  ln_kernel<<<Nq / 8, 256, 0, stream>>>(x32, ln3w, ln3b, xl32, xl16, Nq);
  gemm_nt<<<ggrid(Nq, DFF), 256, 0, stream>>>(xl16, W116, b1, nullptr, hb16,
                                              nullptr, Nq, D, DFF, 1, 1.0f);
  gemm_nt<<<ggrid(Nq, D), 256, 0, stream>>>(hb16, W216, b2, y32, nullptr, xl32,
                                            Nq, DFF, D, 0, 1.0f);
  ln_kernel<<<Nq / 8, 256, 0, stream>>>(y32, ln4w, ln4b, (float*)d_out, nullptr, Nq);
}
